// GSplatRendererBridge_19945828122651
// MI455X (gfx1250) — matrix operand order repaired and checked by reading, not compiled, not run on hardware
//
#include <hip/hip_runtime.h>
#include <hip/hip_bf16.h>

typedef __attribute__((ext_vector_type(16))) _Float16 v16h;
typedef __attribute__((ext_vector_type(8)))  _Float16 v8h;
typedef __attribute__((ext_vector_type(8)))  float    v8f;
typedef _Float16 half_t;

// Problem constants from the reference
constexpr int N_G = 2048;
constexpr int H   = 256;
constexpr int W   = 256;
constexpr int HW  = H * W;
constexpr int PIX_PER_BLOCK = 128;   // 8 waves * 16 pixels

__global__ __launch_bounds__(256)
void gsplat_wmma_kernel(const float* __restrict__ means,    // [N,3]
                        const float* __restrict__ scales,   // [N,3]
                        const float* __restrict__ opac,     // [N]
                        const float* __restrict__ colors,   // [N,3]
                        float* __restrict__ out)            // [3,H,W]
{
    // Per-gaussian params in LDS (44 KB total).
    __shared__ float  s_pxs[N_G];                 // px * inv_s'
    __shared__ float  s_pys[N_G];                 // py * inv_s'
    __shared__ float  s_inv[N_G];                 // inv_s' = sqrt(log2 e / 2) / sigma
    // Transposed, zero-padded B values: rows 0..3 = {o*r, o*g, o*b, o}[g],
    // row 4 = zeros (read by B-columns 4..15). Column-major -> each lane's
    // 16 chunk elements are 32 contiguous bytes.
    __shared__ alignas(16) half_t s_bt[5 * N_G];

    const int tid = threadIdx.x;

    // ---- per-gaussian prep (8 gaussians/thread, redundant per block) ----
    for (int g = tid; g < N_G; g += 256) {
        float mx = means[g * 3 + 0];
        float my = means[g * 3 + 1];
        float mz = means[g * 3 + 2];
        float scl = (scales[g * 3 + 0] + scales[g * 3 + 1] + scales[g * 3 + 2]) * (1.0f / 3.0f);
        scl = fmaxf(scl, 1e-4f);
        float inv_z = 1.0f / (fabsf(mz) + 1.0f);
        float px = tanhf(mx * inv_z);
        float py = tanhf(my * inv_z);
        float sigma = fminf(fmaxf(scl * inv_z, 0.02f), 0.5f);
        // fold -0.5*log2(e) into inv_s so weight = exp2(-(dx'^2 + dy'^2))
        const float kfold = 0.84932180f;   // sqrt(log2(e)/2)
        float inv_s = kfold / sigma;
        s_inv[g] = inv_s;
        s_pxs[g] = px * inv_s;
        s_pys[g] = py * inv_s;
        float o = opac[g];
        s_bt[0 * N_G + g] = (half_t)(o * colors[g * 3 + 0]);
        s_bt[1 * N_G + g] = (half_t)(o * colors[g * 3 + 1]);
        s_bt[2 * N_G + g] = (half_t)(o * colors[g * 3 + 2]);
        s_bt[3 * N_G + g] = (half_t)o;
        s_bt[4 * N_G + g] = (half_t)0.0f;         // zero row for dead B columns
    }
    __syncthreads();

    const int lane = tid & 31;
    const int wave = tid >> 5;
    const int m    = lane & 15;     // A-matrix row owned by this lane
    const int hi   = lane >> 4;     // half-wave: selects K subgroup
    const int ncol = lane & 15;     // B/D column owned by this lane

    const int pix_base = blockIdx.x * PIX_PER_BLOCK + wave * 16;
    const int p  = pix_base + m;    // pixel whose weights this lane computes
    const int pi = p >> 8;          // image row   (yy varies with row)
    const int pj = p & 255;         // image col   (xx varies with col)
    const float xx = -1.0f + (float)pj * (2.0f / (float)(W - 1));
    const float yy = -1.0f + (float)pi * (2.0f / (float)(H - 1));

    // Branch-free B row pointer: real rows for cols 0..3, zero row otherwise.
    const int brow = (ncol < 4) ? ncol : 4;
    const half_t* __restrict__ bptr = &s_bt[brow * N_G + (hi << 4)];

    v8f acc = {};  // D/C accumulator: 16x16 f32

    for (int kbase = 0; kbase < N_G; kbase += 32) {
        // ---- build A tile: 16 pixels x 32 gaussians of f16 weights ----
        // element e of v16h -> K = e + hi*16   (matches B layout)
        v16h a;
#pragma unroll
        for (int e = 0; e < 16; ++e) {
            int g = kbase + e + (hi << 4);
            float inv_s = s_inv[g];
            float dx = fmaf(xx, inv_s, -s_pxs[g]);
            float dy = fmaf(yy, inv_s, -s_pys[g]);
            float t  = fmaf(dy, dy, dx * dx);
            float wgt = __builtin_amdgcn_exp2f(-t);   // v_exp_f32
            a[e] = (half_t)wgt;
        }
        // ---- B tile: 32 contiguous bytes from this lane's column row ----
        // element e of v16h -> K = e + hi*16   (contiguous order, same as A)
        v8h blo = *(const v8h*)(bptr + kbase);
        v8h bhi = *(const v8h*)(bptr + kbase + 8);
        v16h b = __builtin_shufflevector(blo, bhi,
                                         0, 1, 2, 3, 4, 5, 6, 7,
                                         8, 9, 10, 11, 12, 13, 14, 15);
        // D = A x B + C   (v_wmma_f32_16x16x32_f16)
        acc = __builtin_amdgcn_wmma_f32_16x16x32_f16(
            /*neg_a=*/false, a, /*neg_b=*/false, b,
            /*c_mod=*/(short)0, acc, /*reuse_a=*/false, /*reuse_b=*/false);
    }

    // ---- epilogue: D layout VGPR r -> row (r + hi*8), lane -> column ----
    // Branch-free math (v_rcp_f32 instead of IEEE divide; result is clipped
    // to [0,1] so ~1ulp rcp is well within tolerance). Only the store is
    // lane-predicated.
#pragma unroll
    for (int r = 0; r < 8; ++r) {
        float v = acc[r];
        // denom (column 3) lives in lane (hi?19:3) of this half-wave
        float denom = __shfl(v, (lane & 16) | 3, 32);
        float rinv  = __builtin_amdgcn_rcpf(fmaxf(denom, 1e-5f));
        float val   = fminf(fmaxf(v * rinv, 0.0f), 1.0f);
        if (ncol < 3) {
            int prow = pix_base + r + (hi << 3);
            out[ncol * HW + prow] = val;
        }
    }
}

extern "C" void kernel_launch(void* const* d_in, const int* in_sizes, int n_in,
                              void* d_out, int out_size, void* d_ws, size_t ws_size,
                              hipStream_t stream) {
    (void)in_sizes; (void)n_in; (void)d_ws; (void)ws_size; (void)out_size;
    const float* means   = (const float*)d_in[0];   // [N,3]
    // d_in[1] = quats (unused by the reference renderer)
    const float* scales  = (const float*)d_in[2];   // [N,3]
    const float* opac    = (const float*)d_in[3];   // [N]
    const float* colors  = (const float*)d_in[4];   // [N,3]
    float* out = (float*)d_out;                     // [1,3,H,W]

    dim3 grid(HW / PIX_PER_BLOCK);   // 512 blocks
    dim3 block(256);                 // 8 wave32s
    gsplat_wmma_kernel<<<grid, block, 0, stream>>>(means, scales, opac, colors, out);
}
